// attention_62002147885317
// MI455X (gfx1250) — matrix operand order repaired and checked by reading, not compiled, not run on hardware
//
#include <hip/hip_runtime.h>

#define DEVI __device__ __forceinline__

typedef __bf16 v16bf __attribute__((ext_vector_type(16)));
typedef float  v8f   __attribute__((ext_vector_type(8)));
typedef int    gv4i  __attribute__((vector_size(4 * sizeof(int))));  // matches builtin V4i

constexpr int kB = 2, kS = 2048, kNX = 1024, kNH = 16, kHD = 64;
constexpr int kQKV = 3 * kNX; // 3072

// ---- CDNA5 async global->LDS copy (ASYNCcnt path), guarded fallback ----
#if defined(__has_builtin)
#if __has_builtin(__builtin_amdgcn_global_load_async_to_lds_b128) && \
    __has_builtin(__builtin_amdgcn_s_wait_asynccnt)
#define USE_ASYNC_LDS 1
#endif
#endif
#ifndef USE_ASYNC_LDS
#define USE_ASYNC_LDS 0
#endif

DEVI void async_cp128(const __bf16* g, __bf16* l) {
#if USE_ASYNC_LDS
  __builtin_amdgcn_global_load_async_to_lds_b128(
      (__attribute__((address_space(1))) gv4i*)g,
      (__attribute__((address_space(3))) gv4i*)l, 0, 0);
#else
  *(uint4*)l = *(const uint4*)g;
#endif
}

DEVI void async_wait() {
#if USE_ASYNC_LDS
  __builtin_amdgcn_s_wait_asynccnt(0);
#endif
}

union FragU { uint4 u[2]; v16bf v; };

DEVI v16bf ld_frag(const __bf16* base, int e0, int e1) {
  FragU f;
  f.u[0] = *(const uint4*)(base + e0);
  f.u[1] = *(const uint4*)(base + e1);
  return f.v;
}

DEVI v8f wmma_bf16(v16bf a, v16bf b, v8f c) {
  // D = A*B + C, f32 accum, bf16 inputs: v_wmma_f32_16x16x32_bf16
  return __builtin_amdgcn_wmma_f32_16x16x32_bf16(false, a, false, b, (short)0, c,
                                                 false, false);
}

DEVI v8f vzero() {
  v8f z;
#pragma unroll
  for (int i = 0; i < 8; i++) z[i] = 0.f;
  return z;
}

DEVI unsigned pack2(float a, float b) {
  unsigned short lo = __builtin_bit_cast(unsigned short, (__bf16)a);
  unsigned short hi = __builtin_bit_cast(unsigned short, (__bf16)b);
  return (unsigned)lo | ((unsigned)hi << 16);
}

// ---------------------------------------------------------------------------
// GEMM: C[M,N] = A[M,K] @ W[K,N] + bias[N]
// A: fp32 (converted while staging) or bf16 (async direct-to-LDS), W: fp32
// (converted + transposed). Block tile 128x128, K-stage 64, 8 waves (4x2),
// wave tile 32x64.
// ---------------------------------------------------------------------------
template <typename AT, typename OT>
__global__ __launch_bounds__(256) void gemm_bf16_kernel(
    const AT* __restrict__ A, const float* __restrict__ W,
    const float* __restrict__ bias, OT* __restrict__ C, int M, int N, int K) {
  __shared__ __attribute__((aligned(16))) __bf16 Alds[128][72];  // [m][k]
  __shared__ __attribute__((aligned(16))) __bf16 Blds[128][72];  // [n][k]

  const int t = threadIdx.x;
  const int lane = t & 31;
  const int wid = t >> 5;
  const int wm = wid >> 1, wn = wid & 1;
  const int hg = lane >> 4, l16 = lane & 15;
  const int mblk = blockIdx.y * 128;
  const int nblk = blockIdx.x * 128;

  v8f acc[2][4];
#pragma unroll
  for (int i = 0; i < 2; i++)
#pragma unroll
    for (int j = 0; j < 4; j++) acc[i][j] = vzero();

  const int am = t >> 1;            // A stage: row 0..127
  const int aks = (t & 1) * 32;     // k sub-chunk
  const int wk = t >> 2;            // W stage: k row 0..63
  const int wns = (t & 3) * 32;     // n sub-chunk

  for (int k0 = 0; k0 < K; k0 += 64) {
    // ---- stage A tile (128 x 64) -> bf16 LDS ----
    {
      const AT* src = A + (size_t)(mblk + am) * K + k0 + aks;
      if constexpr (__is_same(AT, float)) {
        const float4* s4 = (const float4*)src;
        unsigned* d = (unsigned*)&Alds[am][aks];
#pragma unroll
        for (int i = 0; i < 8; i++) {
          float4 f = s4[i];
          d[2 * i]     = pack2(f.x, f.y);
          d[2 * i + 1] = pack2(f.z, f.w);
        }
      } else {
        __bf16* dst = &Alds[am][aks];
#pragma unroll
        for (int i = 0; i < 4; i++) async_cp128((const __bf16*)src + 8 * i, dst + 8 * i);
      }
    }
    // ---- stage W tile (64 x 128), transposed -> Blds[n][k] ----
    {
      const float4* s4 = (const float4*)(W + (size_t)(k0 + wk) * N + nblk + wns);
#pragma unroll
      for (int i = 0; i < 8; i++) {
        float4 f = s4[i];
        Blds[wns + 4 * i + 0][wk] = (__bf16)f.x;
        Blds[wns + 4 * i + 1][wk] = (__bf16)f.y;
        Blds[wns + 4 * i + 2][wk] = (__bf16)f.z;
        Blds[wns + 4 * i + 3][wk] = (__bf16)f.w;
      }
    }
    // prefetch next K-stage while this one computes
    if (k0 + 64 < K) {
      __builtin_prefetch(A + (size_t)(mblk + am) * K + k0 + 64 + aks, 0, 1);
      __builtin_prefetch(W + (size_t)(k0 + 64 + wk) * N + nblk + wns, 0, 1);
    }
    async_wait();
    __syncthreads();

#pragma unroll
    for (int kki = 0; kki < 2; kki++) {
      const int kk = kki * 32;
      v16bf af[2], bfv[4];
#pragma unroll
      for (int mi = 0; mi < 2; mi++)
        af[mi] = ld_frag(&Alds[wm * 32 + mi * 16 + l16][0],
                         kk + 8 * hg, kk + 16 + 8 * hg);
#pragma unroll
      for (int ni = 0; ni < 4; ni++)
        bfv[ni] = ld_frag(&Blds[wn * 64 + ni * 16 + l16][0],
                          kk + 8 * hg, kk + 16 + 8 * hg);
#pragma unroll
      for (int mi = 0; mi < 2; mi++)
#pragma unroll
        for (int ni = 0; ni < 4; ni++)
          acc[mi][ni] = wmma_bf16(af[mi], bfv[ni], acc[mi][ni]);
    }
    __syncthreads();
  }

  // ---- epilogue: bias add + store ----
#pragma unroll
  for (int mi = 0; mi < 2; mi++) {
#pragma unroll
    for (int ni = 0; ni < 4; ni++) {
      const int col = nblk + wn * 64 + ni * 16 + l16;
      const float bv = bias[col];
#pragma unroll
      for (int r = 0; r < 8; r++) {
        const int row = mblk + wm * 32 + mi * 16 + 8 * hg + r;
        const float v = acc[mi][ni][r] + bv;
        C[(size_t)row * N + col] = (OT)v;
      }
    }
  }
}

// ---------------------------------------------------------------------------
// Causal flash attention.
// Grid: B*NH*(S/64) blocks of 128 threads (4 waves). Wave owns a 16-row Q
// strip; key tiles of 64 staged into LDS (K via async direct-to-LDS, V
// transposed manually); online softmax in fp32 registers.
// qkv layout: [B*S][3072] bf16 with Q|K|V at col offsets 0|1024|2048,
// head h at +h*64.  out: merged heads [B*S][1024] bf16.
// ---------------------------------------------------------------------------
__global__ __launch_bounds__(128) void flash_attn_kernel(
    const __bf16* __restrict__ qkv, __bf16* __restrict__ outm) {
  __shared__ __attribute__((aligned(16))) __bf16 Klds[64][72];      // [key][hd]
  __shared__ __attribute__((aligned(16))) __bf16 Vlds[64][72];      // [hd][key]
  __shared__ __attribute__((aligned(16))) __bf16 Plds[4][16][72];   // per-wave P

  const int qt = blockIdx.x & 31;         // S/64 = 32 q-tiles
  const int h  = (blockIdx.x >> 5) & 15;  // head
  const int b  = blockIdx.x >> 9;         // batch

  const int t = threadIdx.x;
  const int lane = t & 31;
  const int w = t >> 5;  // wave 0..3
  const int hg = lane >> 4, l16 = lane & 15;

  const int qbase = qt * 64 + w * 16;

  // Q strip fragments (16 rows x 64 hd), kept in registers for whole kernel
  v16bf qf[2];
  {
    const __bf16* qrow = qkv + (size_t)(b * kS + qbase + l16) * kQKV + h * kHD;
#pragma unroll
    for (int kki = 0; kki < 2; kki++)
      qf[kki] = ld_frag(qrow, kki * 32 + 8 * hg, kki * 32 + 16 + 8 * hg);
  }

  float mrow[8], lrow[8];
  v8f Oacc[4];
#pragma unroll
  for (int r = 0; r < 8; r++) { mrow[r] = -3.0e38f; lrow[r] = 0.f; }
#pragma unroll
  for (int d = 0; d < 4; d++) Oacc[d] = vzero();

  const int skey = t >> 1;          // staging: key row 0..63
  const int shs  = (t & 1) * 32;    // hd sub-chunk

  for (int kt = 0; kt <= qt; kt++) {
    // ---- stage K (async direct-to-LDS) and V (transposed) tiles ----
    {
      const size_t base =
          (size_t)(b * kS + kt * 64 + skey) * kQKV + h * kHD + shs;
      const __bf16* sk = qkv + base + kNX;   // K region
      __bf16* dk = &Klds[skey][shs];
#pragma unroll
      for (int i = 0; i < 4; i++) async_cp128(sk + 8 * i, dk + 8 * i);

      const uint4* sv = (const uint4*)(qkv + base + 2 * kNX);  // V region
      union { uint4 u[4]; __bf16 e[32]; } vb;
#pragma unroll
      for (int i = 0; i < 4; i++) vb.u[i] = sv[i];
#pragma unroll
      for (int i = 0; i < 32; i++) Vlds[shs + i][skey] = vb.e[i];

      if (kt < qt) {  // prefetch next key tile (K and V regions)
        __builtin_prefetch(qkv + base + (size_t)64 * kQKV + kNX, 0, 1);
        __builtin_prefetch(qkv + base + (size_t)64 * kQKV + 2 * kNX, 0, 1);
      }
    }
    async_wait();
    __syncthreads();

    // ---- S = Q @ K^T (K rows are B-matrix columns directly) ----
    v8f sacc[4];
#pragma unroll
    for (int nt = 0; nt < 4; nt++) {
      sacc[nt] = vzero();
#pragma unroll
      for (int kki = 0; kki < 2; kki++) {
        const int kk = kki * 32;
        v16bf kf = ld_frag(&Klds[nt * 16 + l16][0],
                           kk + 8 * hg, kk + 16 + 8 * hg);
        sacc[nt] = wmma_bf16(qf[kki], kf, sacc[nt]);
      }
    }

    // ---- scale 1/sqrt(64) + causal mask (masked := -10000, as reference) ----
    const bool diag = (kt == qt);
#pragma unroll
    for (int nt = 0; nt < 4; nt++) {
#pragma unroll
      for (int r = 0; r < 8; r++) {
        float sv = sacc[nt][r] * 0.125f;
        if (diag) {
          const int kg = kt * 64 + nt * 16 + l16;
          const int qg = qbase + 8 * hg + r;
          if (kg > qg) sv = -10000.0f;
        }
        sacc[nt][r] = sv;
      }
    }

    // ---- online softmax (rows live in one 16-lane half; xor 8,4,2,1) ----
#pragma unroll
    for (int r = 0; r < 8; r++) {
      float tm = sacc[0][r];
#pragma unroll
      for (int nt = 1; nt < 4; nt++) tm = fmaxf(tm, sacc[nt][r]);
#pragma unroll
      for (int m = 8; m >= 1; m >>= 1) tm = fmaxf(tm, __shfl_xor(tm, m, 32));
      const float mn = fmaxf(mrow[r], tm);
      const float corr = __expf(mrow[r] - mn);
      mrow[r] = mn;
      float rs = 0.f;
#pragma unroll
      for (int nt = 0; nt < 4; nt++) {
        const float pv = __expf(sacc[nt][r] - mn);
        sacc[nt][r] = pv;
        rs += pv;
      }
#pragma unroll
      for (int m = 8; m >= 1; m >>= 1) rs += __shfl_xor(rs, m, 32);
      lrow[r] = lrow[r] * corr + rs;
#pragma unroll
      for (int d = 0; d < 4; d++) Oacc[d][r] = Oacc[d][r] * corr;
    }

    // ---- P: C-layout regs -> row-major LDS (also converts to bf16) ----
#pragma unroll
    for (int nt = 0; nt < 4; nt++)
#pragma unroll
      for (int r = 0; r < 8; r++)
        Plds[w][8 * hg + r][nt * 16 + l16] = (__bf16)sacc[nt][r];
    __syncthreads();

    // ---- O += P @ V ----
#pragma unroll
    for (int d = 0; d < 4; d++) {
#pragma unroll
      for (int kki = 0; kki < 2; kki++) {
        const int kk = kki * 32;
        v16bf pf = ld_frag(&Plds[w][l16][0], kk + 8 * hg, kk + 16 + 8 * hg);
        v16bf vf = ld_frag(&Vlds[d * 16 + l16][0],
                           kk + 8 * hg, kk + 16 + 8 * hg);
        Oacc[d] = wmma_bf16(pf, vf, Oacc[d]);
      }
    }
    __syncthreads();
  }

  // ---- normalize and write merged heads (bf16, feeds projection GEMM) ----
#pragma unroll
  for (int d = 0; d < 4; d++) {
#pragma unroll
    for (int r = 0; r < 8; r++) {
      const float ov = Oacc[d][r] / lrow[r];
      const int row = qbase + 8 * hg + r;
      outm[(size_t)(b * kS + row) * kNX + h * kHD + d * 16 + l16] = (__bf16)ov;
    }
  }
}

// ---------------------------------------------------------------------------
extern "C" void kernel_launch(void* const* d_in, const int* in_sizes, int n_in,
                              void* d_out, int out_size, void* d_ws,
                              size_t ws_size, hipStream_t stream) {
  (void)in_sizes; (void)n_in; (void)out_size; (void)ws_size;
  const float* x      = (const float*)d_in[0];
  const float* w_attn = (const float*)d_in[1];
  const float* b_attn = (const float*)d_in[2];
  const float* w_proj = (const float*)d_in[3];
  const float* b_proj = (const float*)d_in[4];
  float* out = (float*)d_out;

  const int M = kB * kS;  // 4096
  __bf16* qkv  = (__bf16*)d_ws;                                   // 25.2 MB
  __bf16* attn = (__bf16*)((char*)d_ws + (size_t)M * kQKV * 2);   //  8.4 MB

  // 1) QKV = x @ w_attn + b_attn  (fp32 in, bf16 out)
  gemm_bf16_kernel<float, __bf16>
      <<<dim3(kQKV / 128, M / 128), dim3(256), 0, stream>>>(
          x, w_attn, b_attn, qkv, M, kQKV, kNX);

  // 2) causal flash attention per (b, h, 64-row q tile)
  flash_attn_kernel<<<dim3(kB * kNH * (kS / 64)), dim3(128), 0, stream>>>(
      qkv, attn);

  // 3) out = attn @ w_proj + b_proj  (bf16 in, fp32 out)
  gemm_bf16_kernel<__bf16, float>
      <<<dim3(kNX / 128, M / 128), dim3(256), 0, stream>>>(
          attn, w_proj, b_proj, out, M, kNX, kNX);
}
